// TransformerBlock_30107720745796
// MI455X (gfx1250) — compile-verified, matrix operand order checked by reading, not run on hardware
//
#include <hip/hip_runtime.h>
#include <math.h>

// ---------------------------------------------------------------------------
// MI455X (gfx1250) transformer block.
// - All GEMMs: v_wmma_f32_16x16x32_bf16 (bf16 A/B, fp32 accumulate),
//   wave tile 32x64 (8 WMMA / 32-K step), block tile 128x128.
// - A/K tile staging: global_load_async_to_lds_b128 (ASYNCcnt path).
// - Attention: fused flash-style, online softmax, no NxN materialization.
// ---------------------------------------------------------------------------

typedef __attribute__((ext_vector_type(16))) __bf16 v16bf;
typedef __attribute__((ext_vector_type(8)))  float  v8f;

union Frag { v16bf v; unsigned u[8]; };

__device__ __forceinline__ unsigned short f32_to_bf16(float f) {
    union { float f; unsigned u; } x; x.f = f;
    unsigned u = x.u;
    unsigned r = u + 0x7FFFu + ((u >> 16) & 1u);  // round-to-nearest-even
    return (unsigned short)(r >> 16);
}

// Even k-base of dword v (0..7) for the 16-bit A-matrix 16x32 layout
// (ISA 7.12.2): VGPR v holds K pair {k,k+1}; lanes 16..31 add +8.
__device__ __forceinline__ int kpair(int v, int half) {
    return ((v >= 4) ? 16 : 0) + ((v & 3) * 2) + half * 8;
}

// 16x32 bf16 A-style fragment from a row-major [16+][ldh] tile (ldh even).
// For B fragments pass the K-major (transposed) tile Bt[n][k].
__device__ __forceinline__ v16bf load_frag(const unsigned short* __restrict__ base,
                                           int lane, int ldh) {
    int m = lane & 15, half = lane >> 4;
    const unsigned short* p = base + m * ldh;
    Frag f;
#pragma unroll
    for (int v = 0; v < 8; ++v)
        f.u[v] = *(const unsigned*)(p + kpair(v, half));
    return f.v;
}

__device__ __forceinline__ v8f wmma_bf16f32(v16bf a, v16bf b, v8f c) {
    return __builtin_amdgcn_wmma_f32_16x16x32_bf16(
        false, a, false, b, (short)0, c, false, false);
}

__device__ __forceinline__ float gelu_exact(float x) {
    return 0.5f * x * (1.0f + erff(x * 0.70710678118654752f));
}

// Generic pointer to LDS carries the wave-relative LDS byte address in its
// low 32 bits (ISA 10.2).
__device__ __forceinline__ unsigned lds_addr32(const void* p) {
    return (unsigned)(unsigned long long)p;
}

// Async global -> LDS 16B copy (per active lane). ASYNCcnt-tracked.
__device__ __forceinline__ void async_ld_b128(unsigned ldsoff, const void* g) {
    asm volatile("global_load_async_to_lds_b128 %0, %1, off"
                 :: "v"(ldsoff), "v"(g) : "memory");
}
__device__ __forceinline__ void wait_async0() {
    asm volatile("s_wait_asynccnt 0" ::: "memory");
}

// ---------------------------------------------------------------------------
// fp32 -> bf16 conversion (weights)
// ---------------------------------------------------------------------------
__global__ void cvt_bf16_kernel(const float* __restrict__ in,
                                unsigned short* __restrict__ out, int n) {
    int i = blockIdx.x * blockDim.x + threadIdx.x;
    if (i < n) out[i] = f32_to_bf16(in[i]);
}

// ---------------------------------------------------------------------------
// LayerNorm over D=512, one row per block (256 threads = 8 waves), bf16 out.
// ---------------------------------------------------------------------------
__global__ __launch_bounds__(256)
void layernorm_bf16_kernel(const float* __restrict__ x,
                           const float* __restrict__ g,
                           const float* __restrict__ b,
                           unsigned short* __restrict__ out) {
    const int D = 512;
    int row = blockIdx.x, tid = threadIdx.x;
    const float* xr = x + (size_t)row * D;
    float v0 = xr[tid], v1 = xr[tid + 256];
    float s = v0 + v1, s2 = v0 * v0 + v1 * v1;
#pragma unroll
    for (int off = 16; off > 0; off >>= 1) {
        s  += __shfl_xor(s,  off, 32);
        s2 += __shfl_xor(s2, off, 32);
    }
    __shared__ float ps[8], ps2[8];
    int lane = tid & 31, wave = tid >> 5;
    if (lane == 0) { ps[wave] = s; ps2[wave] = s2; }
    __syncthreads();
    if (wave == 0) {
        float a  = (lane < 8) ? ps[lane]  : 0.0f;
        float a2 = (lane < 8) ? ps2[lane] : 0.0f;
#pragma unroll
        for (int off = 4; off > 0; off >>= 1) {
            a  += __shfl_xor(a,  off, 32);
            a2 += __shfl_xor(a2, off, 32);
        }
        if (lane == 0) { ps[0] = a; ps2[0] = a2; }
    }
    __syncthreads();
    float mean = ps[0] * (1.0f / D);
    float var  = ps2[0] * (1.0f / D) - mean * mean;
    float inv  = rsqrtf(var + 1e-6f);
    out[(size_t)row * D + tid]       = f32_to_bf16((v0 - mean) * inv * g[tid]       + b[tid]);
    out[(size_t)row * D + tid + 256] = f32_to_bf16((v1 - mean) * inv * g[tid + 256] + b[tid + 256]);
}

// ---------------------------------------------------------------------------
// GEMM: C[M,N] = A[M,K](bf16) @ B[K,N](bf16)  (+bias, +gelu, +residual)
// 256 threads = 8 waves (4M x 2N), block tile 128x128, wave tile 32x64,
// K step 32. M%128==0, N%128==0, K%32==0 (true for all uses here).
// A tile staged with global_load_async_to_lds_b128; B tile staged transposed.
// ---------------------------------------------------------------------------
#define GT_BM 128
#define GT_BN 128
#define GT_KT 32
#define A_LD  40   // ushorts; 80B row stride (16B aligned for async b128)
#define B_LD  34

template <bool HAS_BIAS, int ACT, bool HAS_RES, bool OUT_F, bool OUT_B>
__global__ __launch_bounds__(256)
void gemm_bf16_kernel(const unsigned short* __restrict__ A,
                      const unsigned short* __restrict__ B,
                      int M, int N, int K,
                      const float* __restrict__ bias,
                      const float* __restrict__ residual,
                      float* __restrict__ outF,
                      unsigned short* __restrict__ outB) {
    __shared__ __align__(16) unsigned short As[GT_BM][A_LD];
    __shared__ __align__(16) unsigned short Bs[GT_BN][B_LD];  // Bt[n][k]

    int tid = threadIdx.x;
    int lane = tid & 31, wave = tid >> 5;
    int wm = wave & 3, wn = wave >> 2;     // 4 (M) x 2 (N) waves
    int bm = blockIdx.x * GT_BM;
    int bn = blockIdx.y * GT_BN;

    v8f c[2][4] = {};

    for (int k0 = 0; k0 < K; k0 += GT_KT) {
        __syncthreads();  // all waves done reading previous tiles
        // --- stage A: 128x32 bf16 via async b128; 2 transfers / thread ---
        {
            int row = tid >> 1, seg2 = (tid & 1) * 2;
            const unsigned short* ga = A + (size_t)(bm + row) * K + k0;
#pragma unroll
            for (int i = 0; i < 2; ++i) {
                int chunk = seg2 + i;  // 0..3, 8 bf16 each
                async_ld_b128(lds_addr32(&As[row][chunk * 8]), ga + chunk * 8);
            }
        }
        // --- stage Bt (transposed): Bt[n][k] = B[(k0+k)*N + bn+n] ---
        {
            int n = tid >> 1, kp = (tid & 1) * 16;
#pragma unroll
            for (int i = 0; i < 16; ++i)
                Bs[n][kp + i] = B[(size_t)(k0 + kp + i) * N + bn + n];
        }
        wait_async0();
        __syncthreads();

        v16bf af[2], bfr[4];
#pragma unroll
        for (int i = 0; i < 2; ++i)
            af[i] = load_frag(&As[wm * 32 + i * 16][0], lane, A_LD);
#pragma unroll
        for (int j = 0; j < 4; ++j)
            bfr[j] = load_frag(&Bs[wn * 64 + j * 16][0], lane, B_LD);
#pragma unroll
        for (int i = 0; i < 2; ++i)
#pragma unroll
            for (int j = 0; j < 4; ++j)
                c[i][j] = wmma_bf16f32(af[i], bfr[j], c[i][j]);
    }

    // epilogue: C/D layout — lane half selects M+8, N = lane&15 (ISA 7.12.2)
    int half = lane >> 4, nl = lane & 15;
#pragma unroll
    for (int i = 0; i < 2; ++i) {
#pragma unroll
        for (int j = 0; j < 4; ++j) {
            int gn = bn + wn * 64 + j * 16 + nl;
            float bv = HAS_BIAS ? bias[gn] : 0.0f;
#pragma unroll
            for (int r = 0; r < 8; ++r) {
                int gm = bm + wm * 32 + i * 16 + r + half * 8;
                float val = c[i][j][r] + bv;
                if (ACT == 1) val = gelu_exact(val);
                size_t idx = (size_t)gm * N + gn;
                if (HAS_RES) val += residual[idx];
                if (OUT_F) outF[idx] = val;
                if (OUT_B) outB[idx] = f32_to_bf16(val);
            }
        }
    }
}

// ---------------------------------------------------------------------------
// Fused flash attention.  Q,K,V bf16 [B*N, 512] (head h = cols h*64..h*64+63).
// grid = (N/128, H, B); block 256 = 8 waves; wave owns 16 query rows.
// Key chunks of 32; S via 4 WMMAs; online softmax; O += P@V via 4 WMMAs.
// K tile staged with async b128; V tile staged transposed (scalar).
// ---------------------------------------------------------------------------
#define K_LD 72   // ushorts; 144B row stride (16B aligned for async b128)

__global__ __launch_bounds__(256)
void attn_kernel(const unsigned short* __restrict__ Q,
                 const unsigned short* __restrict__ Km,
                 const unsigned short* __restrict__ V,
                 unsigned short* __restrict__ O) {
    const int Nn = 4096, D = 512, Dh = 64;
    int h = blockIdx.y, b = blockIdx.z;
    int tid = threadIdx.x, lane = tid & 31, wave = tid >> 5;
    int q0 = blockIdx.x * 128 + wave * 16;
    size_t base = (size_t)b * Nn * D + (size_t)h * Dh;

    __shared__ __align__(16) unsigned short Ks[32][K_LD];      // Ks[key][d]
    __shared__ __align__(16) unsigned short Vt[Dh][32 + 2];    // Vt[d][key]
    __shared__ __align__(16) unsigned short Ps[8][16][32 + 2]; // per-wave P

    // Q fragments (A-layout, two K-chunks of the Dh=64 contraction)
    Frag qa[2];
    {
        int m = lane & 15, half = lane >> 4;
        const unsigned short* qrow = Q + base + (size_t)(q0 + m) * D;
#pragma unroll
        for (int kk = 0; kk < 2; ++kk)
#pragma unroll
            for (int v = 0; v < 8; ++v)
                qa[kk].u[v] = *(const unsigned*)(qrow + kk * 32 + kpair(v, half));
    }

    v8f oacc[4] = {};
    float mrun[8], lrun[8];
#pragma unroll
    for (int r = 0; r < 8; ++r) { mrun[r] = -1e30f; lrun[r] = 0.0f; }

    const float scale = 0.125f;  // Dh^-0.5

    for (int kc = 0; kc < Nn; kc += 32) {
        __syncthreads();  // protect Ks/Vt reuse
        // --- stage K chunk (32x64 bf16) via async b128; 1 transfer / thread
        {
            int key = tid >> 3, seg = tid & 7;
            async_ld_b128(lds_addr32(&Ks[key][seg * 8]),
                          Km + base + (size_t)(kc + key) * D + seg * 8);
        }
        // --- stage Vt transposed: Vt[d][key] = V[key][d] ---
        {
            int key = tid >> 3, dp = (tid & 7) * 8;
            const unsigned short* gv = V + base + (size_t)(kc + key) * D + dp;
#pragma unroll
            for (int i = 0; i < 8; ++i) Vt[dp + i][key] = gv[i];
        }
        wait_async0();
        __syncthreads();

        // S = Q @ K^T   (Bt[n][k] = K[n][k] -> Ks rows used directly)
        v8f s0 = {}, s1 = {};
        s0 = wmma_bf16f32(qa[0].v, load_frag(&Ks[0][0],   lane, K_LD), s0);
        s0 = wmma_bf16f32(qa[1].v, load_frag(&Ks[0][32],  lane, K_LD), s0);
        s1 = wmma_bf16f32(qa[0].v, load_frag(&Ks[16][0],  lane, K_LD), s1);
        s1 = wmma_bf16f32(qa[1].v, load_frag(&Ks[16][32], lane, K_LD), s1);

        // online softmax; 16-lane shuffle groups match C-layout M halves
        float alpha[8];
#pragma unroll
        for (int r = 0; r < 8; ++r) {
            float a0 = s0[r] * scale, a1 = s1[r] * scale;
            float mx = fmaxf(a0, a1);
#pragma unroll
            for (int off = 8; off > 0; off >>= 1)
                mx = fmaxf(mx, __shfl_xor(mx, off, 16));
            float mnew = fmaxf(mrun[r], mx);
            alpha[r] = __expf(mrun[r] - mnew);
            float p0 = __expf(a0 - mnew), p1 = __expf(a1 - mnew);
            s0[r] = p0; s1[r] = p1;
            float rs = p0 + p1;
#pragma unroll
            for (int off = 8; off > 0; off >>= 1)
                rs += __shfl_xor(rs, off, 16);
            lrun[r] = lrun[r] * alpha[r] + rs;
            mrun[r] = mnew;
        }
#pragma unroll
        for (int r = 0; r < 8; ++r) {
            oacc[0][r] *= alpha[r]; oacc[1][r] *= alpha[r];
            oacc[2][r] *= alpha[r]; oacc[3][r] *= alpha[r];
        }

        // P: C-layout -> LDS -> A-layout (per-wave region, DS in-order)
        {
            int hh = lane >> 4, nl = lane & 15;
#pragma unroll
            for (int r = 0; r < 8; ++r) {
                int m = r + hh * 8;
                Ps[wave][m][nl]      = f32_to_bf16(s0[r]);
                Ps[wave][m][16 + nl] = f32_to_bf16(s1[r]);
            }
        }
        Frag pa;
        {
            int m = lane & 15, hh = lane >> 4;
            const unsigned short* pr = &Ps[wave][m][0];
#pragma unroll
            for (int v = 0; v < 8; ++v)
                pa.u[v] = *(const unsigned*)(pr + kpair(v, hh));
        }

        // O += P @ V   (Bt[n][k] = V[k][n] -> Vt)
        oacc[0] = wmma_bf16f32(pa.v, load_frag(&Vt[0][0],  lane, 34), oacc[0]);
        oacc[1] = wmma_bf16f32(pa.v, load_frag(&Vt[16][0], lane, 34), oacc[1]);
        oacc[2] = wmma_bf16f32(pa.v, load_frag(&Vt[32][0], lane, 34), oacc[2]);
        oacc[3] = wmma_bf16f32(pa.v, load_frag(&Vt[48][0], lane, 34), oacc[3]);
    }

    // normalize + store context (bf16)
    int half = lane >> 4, nl = lane & 15;
#pragma unroll
    for (int r = 0; r < 8; ++r) {
        float invl = 1.0f / lrun[r];
        size_t rowb = base + (size_t)(q0 + r + half * 8) * D;
        O[rowb + 0  + nl] = f32_to_bf16(oacc[0][r] * invl);
        O[rowb + 16 + nl] = f32_to_bf16(oacc[1][r] * invl);
        O[rowb + 32 + nl] = f32_to_bf16(oacc[2][r] * invl);
        O[rowb + 48 + nl] = f32_to_bf16(oacc[3][r] * invl);
    }
}

// ---------------------------------------------------------------------------
// Host-side launch sequence
// ---------------------------------------------------------------------------
extern "C" void kernel_launch(void* const* d_in, const int* in_sizes, int n_in,
                              void* d_out, int out_size, void* d_ws, size_t ws_size,
                              hipStream_t stream) {
    (void)in_sizes; (void)n_in; (void)out_size; (void)ws_size;

    const int M = 2 * 4096;     // tokens
    const int D = 512;
    const int MLP = 2048;

    const float* x      = (const float*)d_in[0];
    const float* Wq     = (const float*)d_in[1];
    const float* Wk     = (const float*)d_in[2];
    const float* Wv     = (const float*)d_in[3];
    const float* Wo     = (const float*)d_in[4];
    const float* W1     = (const float*)d_in[5];
    const float* b1     = (const float*)d_in[6];
    const float* W2     = (const float*)d_in[7];
    const float* b2     = (const float*)d_in[8];
    const float* ln_a_g = (const float*)d_in[9];
    const float* ln_a_b = (const float*)d_in[10];
    const float* ln_m_g = (const float*)d_in[11];
    const float* ln_m_b = (const float*)d_in[12];

    char* ws = (char*)d_ws;
    size_t off = 0;
    auto alloc = [&](size_t bytes) -> void* {
        void* p = ws + off;
        off += (bytes + 255) & ~(size_t)255;
        return p;
    };

    unsigned short* wq_b = (unsigned short*)alloc((size_t)D * D * 2);
    unsigned short* wk_b = (unsigned short*)alloc((size_t)D * D * 2);
    unsigned short* wv_b = (unsigned short*)alloc((size_t)D * D * 2);
    unsigned short* wo_b = (unsigned short*)alloc((size_t)D * D * 2);
    unsigned short* w1_b = (unsigned short*)alloc((size_t)D * MLP * 2);
    unsigned short* w2_b = (unsigned short*)alloc((size_t)MLP * D * 2);
    unsigned short* xln  = (unsigned short*)alloc((size_t)M * D * 2);
    unsigned short* qb   = (unsigned short*)alloc((size_t)M * D * 2);
    unsigned short* kb   = (unsigned short*)alloc((size_t)M * D * 2);
    unsigned short* vb   = (unsigned short*)alloc((size_t)M * D * 2);
    unsigned short* ctx  = (unsigned short*)alloc((size_t)M * D * 2);
    float*          y    = (float*)alloc((size_t)M * D * 4);
    unsigned short* zln  = (unsigned short*)alloc((size_t)M * D * 2);
    unsigned short* hb   = (unsigned short*)alloc((size_t)M * MLP * 2);

    // 1) weights -> bf16
    {
        int n = D * D;
        dim3 g((n + 255) / 256);
        cvt_bf16_kernel<<<g, 256, 0, stream>>>(Wq, wq_b, n);
        cvt_bf16_kernel<<<g, 256, 0, stream>>>(Wk, wk_b, n);
        cvt_bf16_kernel<<<g, 256, 0, stream>>>(Wv, wv_b, n);
        cvt_bf16_kernel<<<g, 256, 0, stream>>>(Wo, wo_b, n);
        int n2 = D * MLP;
        dim3 g2((n2 + 255) / 256);
        cvt_bf16_kernel<<<g2, 256, 0, stream>>>(W1, w1_b, n2);
        cvt_bf16_kernel<<<g2, 256, 0, stream>>>(W2, w2_b, n2);
    }

    // 2) LN (attention) -> bf16
    layernorm_bf16_kernel<<<M, 256, 0, stream>>>(x, ln_a_g, ln_a_b, xln);

    // 3) Q,K,V projections (bf16 out)
    dim3 gqkv(M / GT_BM, D / GT_BN);
    gemm_bf16_kernel<false,0,false,false,true><<<gqkv, 256, 0, stream>>>(
        xln, wq_b, M, D, D, nullptr, nullptr, nullptr, qb);
    gemm_bf16_kernel<false,0,false,false,true><<<gqkv, 256, 0, stream>>>(
        xln, wk_b, M, D, D, nullptr, nullptr, nullptr, kb);
    gemm_bf16_kernel<false,0,false,false,true><<<gqkv, 256, 0, stream>>>(
        xln, wv_b, M, D, D, nullptr, nullptr, nullptr, vb);

    // 4) fused flash attention -> bf16 context
    attn_kernel<<<dim3(4096 / 128, 8, 2), 256, 0, stream>>>(qb, kb, vb, ctx);

    // 5) O projection + residual(x) -> fp32 y
    gemm_bf16_kernel<false,0,true,true,false><<<gqkv, 256, 0, stream>>>(
        ctx, wo_b, M, D, D, nullptr, x, y, nullptr);

    // 6) LN (mlp) -> bf16
    layernorm_bf16_kernel<<<M, 256, 0, stream>>>(y, ln_m_g, ln_m_b, zln);

    // 7) MLP up + bias + exact GELU -> bf16 h
    dim3 gup(M / GT_BM, MLP / GT_BN);
    gemm_bf16_kernel<true,1,false,false,true><<<gup, 256, 0, stream>>>(
        zln, w1_b, M, MLP, D, b1, nullptr, nullptr, hb);

    // 8) MLP down + bias + residual(y) -> fp32 out
    gemm_bf16_kernel<true,0,true,true,false><<<gqkv, 256, 0, stream>>>(
        hb, w2_b, M, D, MLP, b2, y, (float*)d_out, nullptr);
}
